// FNetMultiHead_75505525064290
// MI455X (gfx1250) — matrix operand order repaired and checked by reading, not compiled, not run on hardware
//
#include <hip/hip_runtime.h>

typedef __attribute__((ext_vector_type(16))) __bf16 v16bf;
typedef __attribute__((ext_vector_type(8)))  float  v8f;

#define TWO_PI 6.28318530717958647692f
#define USE_ASYNC_LDS 1
#define LDSK 72   // LDS row stride in bf16 elements (144 B = 9x16B, conflict-free)

__device__ __forceinline__ unsigned short f2bf(float f) {
  union { float f; unsigned u; } v; v.f = f;
  unsigned r = v.u + 0x7FFFu + ((v.u >> 16) & 1u);   // round-to-nearest-even
  return (unsigned short)(r >> 16);
}

#if USE_ASYNC_LDS
// gfx1250 async global->LDS copy, 16 bytes per lane. Tracked by ASYNCcnt.
__device__ __forceinline__ void async_copy16(unsigned lds_off, const void* gaddr) {
  asm volatile("global_load_async_to_lds_b128 %0, %1, off"
               :: "v"(lds_off), "v"((unsigned long long)(size_t)gaddr)
               : "memory");
}
__device__ __forceinline__ void wait_async0() {
  asm volatile("s_wait_asynccnt 0" ::: "memory");
}
#endif

// ---------------------------------------------------------------------------
// Kernel 0: CD[c][e] = cos(2*pi*c*e/768), 768x768 f32
// ---------------------------------------------------------------------------
__global__ void cdfill_kernel(float* __restrict__ cd) {
  int idx = blockIdx.x * 256 + threadIdx.x;
  if (idx >= 768 * 768) return;
  int c = idx / 768, e = idx % 768;
  int m = (int)(((long long)c * (long long)e) % 768);
  cd[idx] = __cosf((TWO_PI / 768.0f) * (float)m);
}

// ---------------------------------------------------------------------------
// Kernel 1: stage A — per-head windowed cosine mixing along sequence axis.
// Z[b, t, h*768 + c] (bf16) = sum_j cos(2*pi*(t%w)*j/w) * x[b, blk*w + j, c]
// ---------------------------------------------------------------------------
__global__ void stageA_kernel(const float* __restrict__ x,
                              unsigned short* __restrict__ Z) {
  const int WS[8] = {100, 50, 25, 100, 10, 5, 50, 25};
  extern __shared__ float smem[];
  float* costab = smem;          // up to 100 entries
  float* sx     = smem + 128;    // w * 256 floats

  int j = blockIdx.x;
  int h, w = 0, nblk = 0;
  for (h = 0; h < 8; ++h) {
    w = WS[h]; nblk = 2000 / w;
    int cnt = 8 * nblk * 3;
    if (j < cnt) break;
    j -= cnt;
  }
  if (h >= 8) return;
  int ct  = j % 3; j /= 3;
  int blk = j % nblk;
  int bb  = j / nblk;
  int tid = threadIdx.x;

  for (int k = tid; k < w; k += 256)
    costab[k] = __cosf((TWO_PI / (float)w) * (float)k);

  const float* xp = x + ((size_t)bb * 2000 + (size_t)blk * w) * 768 + ct * 256 + tid;
  for (int r = 0; r < w; ++r)
    sx[r * 256 + tid] = xp[(size_t)r * 768];
  __syncthreads();

  unsigned short* zp =
      Z + ((size_t)bb * 2000 + (size_t)blk * w) * 6144 + h * 768 + ct * 256 + tid;
  for (int i = 0; i < w; ++i) {
    float acc = 0.0f;
    int k = 0;                            // k = (i*r) mod w, incrementally
    for (int r = 0; r < w; ++r) {
      acc += costab[k] * sx[r * 256 + tid];
      k += i; if (k >= w) k -= w;
    }
    zp[(size_t)i * 6144] = f2bf(acc);
  }
}

// ---------------------------------------------------------------------------
// Kernel 2: Gt[o][h*768 + c] = sum_e CD[c][e] * W[o][h*768 + e]   (bf16 out)
// Stored TRANSPOSED (n-major rows, k-contiguous) so the main GEMM's B tile
// is a straight 16B-chunk row copy (no scatter-transpose in the hot loop).
// ---------------------------------------------------------------------------
__global__ void __launch_bounds__(256) gmat_kernel(const float* __restrict__ cd,
                                                   const float* __restrict__ W,
                                                   unsigned short* __restrict__ Gt) {
  __shared__ float sA[64][17];
  __shared__ float sB[16][65];
  int bx = blockIdx.x;
  int h = bx / 144, rem = bx % 144;
  int ctile = (rem / 12) * 64;
  int otile = (rem % 12) * 64;
  int hbase = h * 768;
  int tid = threadIdx.x;
  int tx = tid & 15, ty = tid >> 4;
  float acc[4][4] = {};

  for (int et = 0; et < 768; et += 16) {
#pragma unroll
    for (int q = 0; q < 4; ++q) {
      int l = tid + 256 * q;
      int c = l >> 4, e = l & 15;
      sA[c][e] = cd[(size_t)(ctile + c) * 768 + et + e];
    }
#pragma unroll
    for (int q = 0; q < 4; ++q) {
      int l = tid + 256 * q;
      int o = l >> 4, e = l & 15;
      sB[e][o] = W[(size_t)(otile + o) * 6144 + hbase + et + e];
    }
    __syncthreads();
#pragma unroll
    for (int e = 0; e < 16; ++e)
#pragma unroll
      for (int i = 0; i < 4; ++i)
#pragma unroll
        for (int jj = 0; jj < 4; ++jj)
          acc[i][jj] += sA[ty * 4 + i][e] * sB[e][tx * 4 + jj];
    __syncthreads();
  }
  // transposed store: Gt[n][k], n = output feature, k = h*768 + c
  for (int i = 0; i < 4; ++i)
    for (int jj = 0; jj < 4; ++jj)
      Gt[(size_t)(otile + tx * 4 + jj) * 6144 + hbase + ctile + ty * 4 + i] =
          f2bf(acc[i][jj]);
}

// ---------------------------------------------------------------------------
// Kernel 3: out (16000x768 f32) = Z (16000x6144 bf16) @ Gt^T (bf16) + bias
// v_wmma_f32_16x16x32_bf16; 128x128x64 tiles; 8 wave32 waves (2x4), each wave
// 64x32 = 4x2 accumulators. Double-buffered LDS fed by async global->LDS DMA.
// ---------------------------------------------------------------------------
__global__ void __launch_bounds__(256) wmma_gemm_kernel(
    const unsigned short* __restrict__ Z, const unsigned short* __restrict__ Gt,
    const float* __restrict__ bias, float* __restrict__ out) {
  extern __shared__ unsigned short lds[];
  unsigned short* Abuf = lds;                       // [2][128][LDSK]
  unsigned short* Bbuf = lds + 2 * 128 * LDSK;      // [2][128][LDSK]

  const int tid   = threadIdx.x;
  const int lane  = tid & 31;
  const int wv    = tid >> 5;
  const int waveM = wv >> 2;               // 0..1
  const int waveN = wv & 3;                // 0..3
  const int tileM = blockIdx.y * 128;
  const int tileN = blockIdx.x * 128;

#if USE_ASYNC_LDS
  const unsigned abase = (unsigned)(size_t)Abuf;
  const unsigned bbase = (unsigned)(size_t)Bbuf;
#endif

  // Both tiles are 128 rows x 64 k of bf16: 1024 chunks of 16B, 4 per thread.
  auto issueTile = [&](int kt, int p) {
#pragma unroll
    for (int q = 0; q < 4; ++q) {
      int ch  = tid + 256 * q;
      int row = ch >> 3;
      int col = (ch & 7) << 3;
      unsigned loff = (unsigned)((p * 128 + row) * LDSK + col) * 2u;
      const unsigned short* ga = Z  + (size_t)(tileM + row) * 6144 + kt + col;
      const unsigned short* gb = Gt + (size_t)(tileN + row) * 6144 + kt + col;
#if USE_ASYNC_LDS
      async_copy16(abase + loff, ga);
      async_copy16(bbase + loff, gb);
#else
      *(uint4*)((char*)Abuf + loff) = *(const uint4*)ga;
      *(uint4*)((char*)Bbuf + loff) = *(const uint4*)gb;
#endif
    }
  };

  v8f acc[4][2] = {};
  int p = 0;
  issueTile(0, 0);
#if USE_ASYNC_LDS
  wait_async0();
#endif
  __syncthreads();

  for (int kt = 0; kt < 6144; kt += 64) {
    if (kt + 64 < 6144) issueTile(kt + 64, p ^ 1);

    const unsigned short* As = Abuf + p * 128 * LDSK;
    const unsigned short* Bs = Bbuf + p * 128 * LDSK;
#pragma unroll
    for (int kk = 0; kk < 64; kk += 32) {
      union { v16bf v; uint4 u[2]; } a[4], b[2];
      const int rsel = lane & 15;
      const int kgA  = (lane >> 4) << 3;   // A halves: K 0-7/16-23 vs 8-15/24-31
      const int kgB  = (lane >> 4) << 3;   // B halves: K 0-7/16-23 vs 8-15/24-31
#pragma unroll
      for (int mi = 0; mi < 4; ++mi) {
        const unsigned short* r = As + (waveM * 64 + mi * 16 + rsel) * LDSK;
        a[mi].u[0] = *(const uint4*)(r + kk + kgA);
        a[mi].u[1] = *(const uint4*)(r + kk + 16 + kgA);
      }
#pragma unroll
      for (int ni = 0; ni < 2; ++ni) {
        const unsigned short* r = Bs + (waveN * 32 + ni * 16 + rsel) * LDSK;
        b[ni].u[0] = *(const uint4*)(r + kk + kgB);
        b[ni].u[1] = *(const uint4*)(r + kk + 16 + kgB);
      }
#pragma unroll
      for (int mi = 0; mi < 4; ++mi)
#pragma unroll
        for (int ni = 0; ni < 2; ++ni)
          acc[mi][ni] = __builtin_amdgcn_wmma_f32_16x16x32_bf16(
              false, a[mi].v, false, b[ni].v, (short)0, acc[mi][ni], false, false);
    }
#if USE_ASYNC_LDS
    wait_async0();
#endif
    __syncthreads();
    p ^= 1;
  }

  // store D + bias; C/D layout: VGPR v -> M = (lane>>4)*8 + v, N = lane&15
  const int rsel = lane & 15;
  const int rhi  = (lane >> 4) * 8;
#pragma unroll
  for (int mi = 0; mi < 4; ++mi)
#pragma unroll
    for (int ni = 0; ni < 2; ++ni) {
      int col = tileN + waveN * 32 + ni * 16 + rsel;
      float bv = bias[col];
#pragma unroll
      for (int v = 0; v < 8; ++v) {
        int row = tileM + waveM * 64 + mi * 16 + rhi + v;
        out[(size_t)row * 768 + col] = acc[mi][ni][v] + bv;
      }
    }
}

// ---------------------------------------------------------------------------
extern "C" void kernel_launch(void* const* d_in, const int* in_sizes, int n_in,
                              void* d_out, int out_size, void* d_ws, size_t ws_size,
                              hipStream_t stream) {
  const float* x    = (const float*)d_in[0];   // (8, 2000, 768)
  const float* W    = (const float*)d_in[1];   // (768, 6144)
  const float* bias = (const float*)d_in[2];   // (768,)
  float* out = (float*)d_out;                  // (8, 2000, 768)

  // workspace layout
  unsigned short* Z = (unsigned short*)d_ws;                 // 16000*6144 bf16
  size_t zbytes  = (size_t)16000 * 6144 * 2;                 // 196.6 MB
  float* CD = (float*)((char*)d_ws + zbytes);                // 768*768 f32
  size_t cdbytes = (size_t)768 * 768 * 4;                    // 2.36 MB
  unsigned short* Gt = (unsigned short*)((char*)d_ws + zbytes + cdbytes); // 768*6144 bf16

  // 0) cosine matrix for embedding-axis transform
  cdfill_kernel<<<(768 * 768 + 255) / 256, 256, 0, stream>>>(CD);

  // 1) windowed cosine mixing -> Z (bf16)
  const int WS[8] = {100, 50, 25, 100, 10, 5, 50, 25};
  int totalJobs = 0;
  for (int h = 0; h < 8; ++h) totalJobs += 8 * (2000 / WS[h]) * 3;
  size_t shA = (size_t)(128 + 100 * 256) * sizeof(float);    // ~103 KB dynamic LDS
  stageA_kernel<<<totalJobs, 256, shA, stream>>>(x, Z);

  // 2) Gt = (blockdiag(C_d) @ W^T)^T  (folded weights, bf16, pre-transposed)
  gmat_kernel<<<8 * 144, 256, 0, stream>>>(CD, W, Gt);

  // 3) out = Z @ Gt^T + bias  (WMMA bf16 -> f32, async double-buffered LDS)
  dim3 grid(768 / 128, 16000 / 128);
  size_t shG = (size_t)4 * 128 * LDSK * sizeof(unsigned short); // 73728 B
  wmma_gemm_kernel<<<grid, 256, shG, stream>>>(Z, Gt, bias, out);
}
